// RelativeMultiHeadAttention_25082609009119
// MI455X (gfx1250) — matrix operand order repaired and checked by reading, not compiled, not run on hardware
//
#include <hip/hip_runtime.h>

typedef __attribute__((ext_vector_type(4)))  float  v4f;
typedef __attribute__((ext_vector_type(8)))  float  v8f;
typedef __attribute__((ext_vector_type(16))) float  v16f;
typedef __attribute__((ext_vector_type(8)))  __bf16 v8bf;
typedef __attribute__((ext_vector_type(16))) __bf16 v16bf;

#define DIMD 512
#define HN   8
#define HD   64
#define LL   16
#define NTOK 64
#define NN   1024
#define BF16_MIN_F (-3.3895313892515355e38f)
#define INV_SCALE  0.04419417382415922f  /* 1/sqrt(512) */

__device__ __forceinline__ v16bf cat8(v8bf lo, v8bf hi) {
  return __builtin_shufflevector(lo, hi, 0,1,2,3,4,5,6,7,8,9,10,11,12,13,14,15);
}

/* ---------------- Kernel 1: time projection tp[bl][192] ---------------- */
__global__ void k_tproj(const float* __restrict__ te, const float* __restrict__ Wt,
                        float* __restrict__ tp) {
  int bl = blockIdx.x;            // 0..63  (b*16 + l)
  int j  = threadIdx.x;           // 0..191 (3*hd output channels)
  const v4f* t = (const v4f*)(te + (size_t)bl * DIMD);
  const v4f* w = (const v4f*)(Wt + (size_t)j  * DIMD);
  float acc = 0.f;
  for (int d = 0; d < DIMD / 4; ++d) {
    v4f a = t[d], b = w[d];
    acc += a.x * b.x + a.y * b.y + a.z * b.z + a.w * b.w;
  }
  tp[bl * 192 + j] = acc;
}

/* ------- Kernel 2: QKV projection GEMM (WMMA) + time bias + scatter ------- */
/* grid (24, 4, 64), block 128 (4 waves). Wave = one 16x16 output tile.      */
__global__ __launch_bounds__(128) void k_qkv(
    const float* __restrict__ x, const float* __restrict__ Wq,
    const float* __restrict__ tp,
    __bf16* __restrict__ Q, __bf16* __restrict__ K, __bf16* __restrict__ Vt) {
  int tid  = threadIdx.x;
  int w    = tid >> 5;
  int L    = tid & 31;
  int half = L >> 4;
  int lm   = L & 15;
  int n0   = (blockIdx.x * 4 + w) * 16;   // output channel tile base (0..1520)
  int m0   = blockIdx.y * 16;             // token tile base (0..48)
  int bl   = blockIdx.z;                  // b*16 + l

  const float* xrow = x  + ((size_t)bl * NTOK + (m0 + lm)) * DIMD; // A: row = token
  const float* wrow = Wq + (size_t)(n0 + lm) * DIMD;               // B: col = out ch

  v8f acc = {0.f,0.f,0.f,0.f,0.f,0.f,0.f,0.f};
  for (int k0 = 0; k0 < DIMD; k0 += 32) {
    v8f  xa = *(const v8f*)(xrow + k0 + half * 16);       // A: K = half*16 + 0..7
    v8f  xb = *(const v8f*)(xrow + k0 + half * 16 + 8);   // A: K = half*16+8 + 0..7
    v16f wv = *(const v16f*)(wrow + k0 + half * 16);      // B: K = half*16 + 0..15
    v16bf a  = cat8(__builtin_convertvector(xa, v8bf),
                    __builtin_convertvector(xb, v8bf));
    v16bf bf = __builtin_convertvector(wv, v16bf);
    acc = __builtin_amdgcn_wmma_f32_16x16x32_bf16(false, a, false, bf,
                                                  (short)0, acc, false, false);
  }

  /* section / head are uniform per tile (n0 is 16-aligned, sections 512-aligned) */
  int sec   = n0 >> 9;            // 0=q 1=k 2=v  (wave-uniform)
  int rbase = n0 & 511;
  int head  = rbase >> 6;         // wave-uniform
  int c     = (rbase & 63) + lm;  // channel within head, per lane
  float tb  = tp[bl * 192 + sec * 64 + c];
  int bb = bl >> 4, ll2 = bl & 15;
  int bh = bb * HN + head;

  if (sec == 2) {
#pragma unroll
    for (int v = 0; v < 8; ++v) {
      int Nidx = ll2 * NTOK + m0 + v + 8 * half;
      Vt[((size_t)bh * HD + c) * NN + Nidx] = (__bf16)(acc[v] + tb);  // transposed
    }
  } else {
    __bf16* dst = (sec == 0) ? Q : K;
#pragma unroll
    for (int v = 0; v < 8; ++v) {
      int Nidx = ll2 * NTOK + m0 + v + 8 * half;
      dst[((size_t)bh * NN + Nidx) * HD + c] = (__bf16)(acc[v] + tb);
    }
  }
}

/* ---- Kernel 3: attention. grid (64, 8, 4), block 256 (8 waves). ---- */
__global__ __launch_bounds__(256) void k_attn(
    const __bf16* __restrict__ Q, const __bf16* __restrict__ Kb,
    const __bf16* __restrict__ Vt,
    const int* __restrict__ pos, const int* __restrict__ mask,
    const float* __restrict__ emb, float* __restrict__ out) {
  __shared__ float sc[16][NN];   // 64 KB score strip for 16 query rows

  int tid  = threadIdx.x;
  int w    = tid >> 5;
  int L    = tid & 31;
  int half = L >> 4;
  int lm   = L & 15;
  int m0   = blockIdx.x * 16;    // query row tile in N
  int h    = blockIdx.y;
  int b    = blockIdx.z;
  int ql   = m0 >> 6;            // query frame
  int bh   = b * HN + h;

  /* Q A-fragments for K-steps 0 and 32 (hd = 64): contiguous 16-elt per half */
  const __bf16* qrow = Q + ((size_t)bh * NN + m0 + lm) * HD;
  v16bf aq0 = cat8(*(const v8bf*)(qrow +      half * 16),
                   *(const v8bf*)(qrow +  8 + half * 16));
  v16bf aq1 = cat8(*(const v8bf*)(qrow + 32 + half * 16),
                   *(const v8bf*)(qrow + 40 + half * 16));

  /* relative-position bias tile [16 rows][16 key frames], per-wave in regs */
  int p = pos[(b * LL + ql) * LL + lm];
  p = min(8, max(-8, p)) + 8;
  const float* erow = emb + p * HD;
  v8f bias = {0.f,0.f,0.f,0.f,0.f,0.f,0.f,0.f};
  {
    v16bf be0 = __builtin_convertvector(*(const v16f*)(erow + half * 16), v16bf);
    v16bf be1 = __builtin_convertvector(*(const v16f*)(erow + 32 + half * 16), v16bf);
    bias = __builtin_amdgcn_wmma_f32_16x16x32_bf16(false, aq0, false, be0,
                                                   (short)0, bias, false, false);
    bias = __builtin_amdgcn_wmma_f32_16x16x32_bf16(false, aq1, false, be1,
                                                   (short)0, bias, false, false);
  }

  /* causal frame mask bits */
  unsigned mbits = 0;
  const int* mrow = mask + (b * LL + ql) * LL;
#pragma unroll
  for (int kl = 0; kl < 16; ++kl)
    if (mrow[kl] != 0) mbits |= (1u << kl);

  /* QK^T scores: wave w covers key tiles w*8 .. w*8+7 */
  for (int i = 0; i < 8; ++i) {
    int kt = w * 8 + i;
    int kl = kt >> 2;                       // key frame (4 tiles per frame)
    const __bf16* krow = Kb + ((size_t)bh * NN + kt * 16 + lm) * HD;
    v16bf bk0 = *(const v16bf*)(krow +      half * 16);
    v16bf bk1 = *(const v16bf*)(krow + 32 + half * 16);
    v8f acc = {0.f,0.f,0.f,0.f,0.f,0.f,0.f,0.f};
    acc = __builtin_amdgcn_wmma_f32_16x16x32_bf16(false, aq0, false, bk0,
                                                  (short)0, acc, false, false);
    acc = __builtin_amdgcn_wmma_f32_16x16x32_bf16(false, aq1, false, bk1,
                                                  (short)0, acc, false, false);
    bool allowed = (mbits >> kl) & 1u;
    int key = kt * 16 + lm;
#pragma unroll
    for (int v = 0; v < 8; ++v) {
      float bv = __shfl(bias[v], (half << 4) | kl, 32); // col kl, same half
      float sv = allowed ? (acc[v] + bv) * INV_SCALE : BF16_MIN_F;
      sc[v + 8 * half][key] = sv;
    }
  }
  __syncthreads();

  /* one-pass softmax: 2 rows per wave, shuffle reductions (wave32) */
  for (int rr = 0; rr < 2; ++rr) {
    int r = w * 2 + rr;
    float mx = -3.4e38f;
    for (int i = 0; i < 32; ++i) mx = fmaxf(mx, sc[r][L + 32 * i]);
    for (int off = 16; off > 0; off >>= 1) mx = fmaxf(mx, __shfl_xor(mx, off, 32));
    float sum = 0.f;
    for (int i = 0; i < 32; ++i) {
      float e = __expf(sc[r][L + 32 * i] - mx);
      sc[r][L + 32 * i] = e;
      sum += e;
    }
    for (int off = 16; off > 0; off >>= 1) sum += __shfl_xor(sum, off, 32);
    float inv = 1.f / sum;
    for (int i = 0; i < 32; ++i) sc[r][L + 32 * i] *= inv;
  }
  __syncthreads();

  /* P @ V: waves 0..3 each own a 16-wide slice of hd, K = 1024 in 32 steps */
  if (w < 4) {
    int vd0 = w * 16;
    const __bf16* vrow = Vt + ((size_t)bh * HD + vd0 + lm) * NN;
    v8f acc = {0.f,0.f,0.f,0.f,0.f,0.f,0.f,0.f};
    for (int kt2 = 0; kt2 < 32; ++kt2) {
      int k0 = kt2 * 32;
      v8f p0 = *(const v8f*)(&sc[lm][k0 + half * 16]);      // A: keys half*16+0..7
      v8f p1 = *(const v8f*)(&sc[lm][k0 + half * 16 + 8]); // A: keys half*16+8+..
      v16bf a  = cat8(__builtin_convertvector(p0, v8bf),
                      __builtin_convertvector(p1, v8bf));
      v16bf bv = *(const v16bf*)(vrow + k0 + half * 16);    // B: keys half*16+0..15
      acc = __builtin_amdgcn_wmma_f32_16x16x32_bf16(false, a, false, bv,
                                                    (short)0, acc, false, false);
    }
#pragma unroll
    for (int v = 0; v < 8; ++v) {
      int m  = m0 + v + 8 * half;     // global query row
      int lf = m >> 6, tok = m & 63;
      int ch = h * HD + vd0 + lm;
      out[(((size_t)b * LL + lf) * NTOK + tok) * DIMD + ch] = acc[v];
    }
  }
}

/* ------------------------------ launcher ------------------------------ */
extern "C" void kernel_launch(void* const* d_in, const int* in_sizes, int n_in,
                              void* d_out, int out_size, void* d_ws, size_t ws_size,
                              hipStream_t stream) {
  const float* x    = (const float*)d_in[0];
  const float* te   = (const float*)d_in[1];
  const int*   pos  = (const int*)d_in[2];
  const int*   mask = (const int*)d_in[3];
  const float* Wq   = (const float*)d_in[4];
  const float* Wt   = (const float*)d_in[5];
  const float* emb  = (const float*)d_in[6];
  float* out = (float*)d_out;

  char* ws = (char*)d_ws;
  float*  tp = (float*)ws;                           /* 64*192*4   = 48 KB  */
  __bf16* Q  = (__bf16*)(ws + 65536);                /* 4 MB bf16           */
  __bf16* K  = (__bf16*)(ws + 65536 + 4194304);      /* 4 MB bf16           */
  __bf16* Vt = (__bf16*)(ws + 65536 + 8388608);      /* 4 MB bf16 (transposed) */

  k_tproj<<<64, 192, 0, stream>>>(te, Wt, tp);
  k_qkv  <<<dim3(24, 4, 64), 128, 0, stream>>>(x, Wq, tp, Q, K, Vt);
  k_attn <<<dim3(64, 8, 4), 256, 0, stream>>>(Q, K, Vt, pos, mask, emb, out);
}
